// MultiScaleGNNBlock_17506286698855
// MI455X (gfx1250) — matrix operand order repaired and checked by reading, not compiled, not run on hardware
//
#include <hip/hip_runtime.h>
#include <hip/hip_bf16.h>

typedef __attribute__((ext_vector_type(16))) _Float16 v16h;
typedef __attribute__((ext_vector_type(4)))  _Float16 v4h;
typedef __attribute__((ext_vector_type(8)))  float    v8f;
typedef __attribute__((ext_vector_type(4)))  float    v4f;

#define NNODES 50000
#define NEDGES 800000
#define NEDGESP (NEDGES + NNODES)   // with self loops
#define NET 7
#define BN_SCALE 0.9999950000375f   // 1/sqrt(1+1e-5)

// ---------------- WMMA fragment layout helpers (CDNA5 16-bit, wave32) ----------
// A matrix 16x32 (MxK): lanes 0-15 hold K {0..7,16..23}, lanes 16-31 hold K {8..15,24..31}
__device__ __forceinline__ int afrag_lane(int m, int k) { return (m & 15) + (((k >> 3) & 1) << 4); }
__device__ __forceinline__ int afrag_idx(int k)         { return (((k >> 4) & 1) << 3) + (k & 7); }
// B matrix 32x16 (KxN): lanes 0-15 hold K {0..7,16..23}, lanes 16-31 hold K {8..15,24..31}
__device__ __forceinline__ int bfrag_lane(int k, int n) { return (n & 15) + (((k >> 3) & 1) << 4); }
__device__ __forceinline__ int bfrag_idx(int k)         { return (((k >> 4) & 1) << 3) + (k & 7); }

__device__ __forceinline__ v8f wmma_f16(v16h a, v16h b, v8f c) {
  return __builtin_amdgcn_wmma_f32_16x16x32_f16(false, a, false, b, (short)0, c, false, false);
}

// ordered-int encoding for float atomicMax
__device__ __forceinline__ unsigned f2ord(float f) {
  unsigned u = __float_as_uint(f);
  return (u & 0x80000000u) ? ~u : (u | 0x80000000u);
}
__device__ __forceinline__ float ord2f(unsigned u) {
  return (u & 0x80000000u) ? __uint_as_float(u & 0x7fffffffu) : __uint_as_float(~u);
}

// ---------------- generic fill ----------------
__global__ void k_fill(unsigned* p, unsigned v, long n) {
  long i = (long)blockIdx.x * blockDim.x + threadIdx.x;
  if (i < n) p[i] = v;
}

// ---------------- pack f32 weight [Nout,K] into f16 B-fragment tiles ----------
// tile (kt,nt) at offset (nt*(Kpad/32)+kt)*512 halves; within tile: lane*16+idx
__global__ void k_pack_w(const float* __restrict__ W, _Float16* __restrict__ out,
                         int Nout, int K, int Kpad) {
  long total = (long)Nout * Kpad;
  for (long i = (long)blockIdx.x * blockDim.x + threadIdx.x; i < total;
       i += (long)gridDim.x * blockDim.x) {
    int n = (int)(i / Kpad), k = (int)(i % Kpad);
    float v = (k < K) ? W[(size_t)n * K + k] : 0.f;
    int kt = k >> 5, nt = n >> 4, kl = k & 31;
    size_t pos = ((size_t)(nt * (Kpad >> 5) + kt)) * 512 + bfrag_lane(kl, n) * 16 + bfrag_idx(kl);
    out[pos] = (_Float16)v;
  }
}

// ---------------- small precompute: edge-att projections, self-loop alpha, bias --
__global__ void k_small_pre(const float* __restrict__ lin_edge, const float* __restrict__ att_edge,
                            const float* __restrict__ gat_ete, const float* __restrict__ lin_out_b,
                            const float* __restrict__ gat_bias,
                            float* __restrict__ w_e, float* __restrict__ t_alpha,
                            float* __restrict__ alpha_self, float* __restrict__ comb_bias) {
  __shared__ float s_we[64], s_ta[28];
  int tid = threadIdx.x;
  if (tid < 64) {                       // w_e[h][d] = sum_c lin_edge[h*32+c][d]*att_edge[h][c]
    int h = tid >> 4, d = tid & 15;
    float s = 0.f;
    for (int c = 0; c < 32; ++c) s += lin_edge[(h * 32 + c) * 16 + d] * att_edge[h * 32 + c];
    s_we[tid] = s; w_e[tid] = s;
  }
  if (tid >= 64 && tid < 92) {          // t_alpha[t][h]
    int i = tid - 64, t = i >> 2, h = i & 3;
    float s = 0.f;
    for (int c = 0; c < 32; ++c) s += gat_ete[t * 128 + h * 32 + c] * att_edge[h * 32 + c];
    s_ta[i] = s; t_alpha[i] = s;
  }
  if (tid < 128) comb_bias[tid] = lin_out_b[tid] + gat_bias[tid];
  __syncthreads();
  if (tid < 4) {                        // self loop: edge_attr = ones, type = NET-1
    float s = 0.f;
    for (int d = 0; d < 16; ++d) s += s_we[tid * 16 + d];
    alpha_self[tid] = s + s_ta[(NET - 1) * 4 + tid];
  }
}

// ---------------- WMMA GEMM: C = epi(A[M,K] @ W^T) ----------------
// block = 128 threads (4 waves), tile 16 rows x 128 cols; wave w owns column
// tiles w and w+4 (2 WMMAs per k-step sharing one A fragment). K multiple of 32.
__global__ void k_gemm(const float* __restrict__ A, int lda,
                       const _Float16* __restrict__ Bf,
                       const float* __restrict__ bias,
                       const float* __restrict__ scale, float scale_c,
                       const float* __restrict__ shift,
                       const float* __restrict__ residual, int ldres,
                       float* __restrict__ C, int ldc, int K, int relu_flag) {
  __shared__ __align__(32) _Float16 sA[512];
  int tid = threadIdx.x, wave = tid >> 5, lane = tid & 31;
  int row0 = blockIdx.x * 16;
  int ktiles = K >> 5;
  int nt0 = blockIdx.y * 8 + wave;             // first n-tile, second is nt0+4
  const _Float16* B0 = Bf + (size_t)nt0 * ktiles * 512;
  const _Float16* B1 = Bf + (size_t)(nt0 + 4) * ktiles * 512;
  // staging map: thread t covers (m = t&15, k = (t>>4)*4 .. +3), one b64 LDS store
  int sm = tid & 15, sk = (tid >> 4) << 2;
  const float* Arow = A + (size_t)(row0 + sm) * lda + sk;
  _Float16* sdst = &sA[afrag_lane(sm, sk) * 16 + afrag_idx(sk)];
  v8f acc0 = {}, acc1 = {};
  for (int kt = 0; kt < ktiles; ++kt) {
    v4f av = *(const v4f*)(Arow + kt * 32);
    *(v4h*)sdst = __builtin_convertvector(av, v4h);
    __syncthreads();
    v16h a  = *(const v16h*)&sA[lane * 16];
    v16h b0 = *(const v16h*)&B0[(size_t)kt * 512 + lane * 16];
    v16h b1 = *(const v16h*)&B1[(size_t)kt * 512 + lane * 16];
    if (kt + 1 < ktiles) {
      __builtin_prefetch(&B0[(size_t)(kt + 1) * 512 + lane * 16], 0, 1);
      __builtin_prefetch(&B1[(size_t)(kt + 1) * 512 + lane * 16], 0, 1);
    }
    acc0 = wmma_f16(a, b0, acc0);
    acc1 = wmma_f16(a, b1, acc1);
    __syncthreads();
  }
  int n0c = blockIdx.y * 128 + wave * 16 + (lane & 15);
  int n1c = n0c + 64;
  int mhi = (lane >> 4) << 3;
#pragma unroll
  for (int r = 0; r < 8; ++r) {
    int row = row0 + r + mhi;
    float v0 = acc0[r], v1 = acc1[r];
    if (bias)  { v0 += bias[n0c];  v1 += bias[n1c]; }
    if (scale) { v0 *= scale[n0c]; v1 *= scale[n1c]; }
    v0 *= scale_c; v1 *= scale_c;
    if (shift) { v0 += shift[n0c]; v1 += shift[n1c]; }
    if (relu_flag) { v0 = v0 > 0.f ? v0 : 0.f; v1 = v1 > 0.f ? v1 : 0.f; }
    if (residual) {
      v0 += residual[(size_t)row * ldres + n0c];
      v1 += residual[(size_t)row * ldres + n1c];
    }
    C[(size_t)row * ldc + n0c] = v0;
    C[(size_t)row * ldc + n1c] = v1;
  }
}

// ---------------- per-node attention logit projections -------------------------
__global__ void k_ai_aj(const float* __restrict__ xsrc, const float* __restrict__ xdst,
                        const float* __restrict__ att_src, const float* __restrict__ att_dst,
                        float* __restrict__ ai, float* __restrict__ aj, int n_nodes) {
  int i = blockIdx.x * blockDim.x + threadIdx.x;
  if (i >= n_nodes * 4) return;
  int n = i >> 2, h = i & 3;
  float si = 0.f, sj = 0.f;
  for (int c = 0; c < 32; ++c) {
    si += xdst[(size_t)n * 128 + h * 32 + c] * att_src[h * 32 + c];
    sj += xsrc[(size_t)n * 128 + h * 32 + c] * att_dst[h * 32 + c];
  }
  ai[i] = si; aj[i] = sj;
}

// ---------------- GAT pass A: logits + segment max -----------------------------
__global__ void k_gat_alpha(const int* __restrict__ ei, const int* __restrict__ et,
                            const float* __restrict__ ea,
                            const float* __restrict__ ai, const float* __restrict__ aj,
                            const float* __restrict__ w_e, const float* __restrict__ t_alpha,
                            const float* __restrict__ alpha_self,
                            float* __restrict__ alpha, unsigned* __restrict__ mmax) {
  int e = blockIdx.x * blockDim.x + threadIdx.x;
  if (e >= NEDGESP) return;
  int s, d;
  float av[4];
  if (e < NEDGES) {
    s = ei[e]; d = ei[NEDGES + e];
    int t = et[e];
    float eav[16];
    const float4* eap = (const float4*)(ea + (size_t)e * 16);
    ((float4*)eav)[0] = eap[0]; ((float4*)eav)[1] = eap[1];
    ((float4*)eav)[2] = eap[2]; ((float4*)eav)[3] = eap[3];
#pragma unroll
    for (int h = 0; h < 4; ++h) {
      float sum = t_alpha[t * 4 + h];
#pragma unroll
      for (int k = 0; k < 16; ++k) sum += eav[k] * w_e[h * 16 + k];
      av[h] = sum;
    }
  } else {
    s = d = e - NEDGES;
#pragma unroll
    for (int h = 0; h < 4; ++h) av[h] = alpha_self[h];
  }
#pragma unroll
  for (int h = 0; h < 4; ++h) {
    float v = ai[d * 4 + h] + aj[s * 4 + h] + av[h];
    v = v > 0.f ? v : 0.2f * v;                  // leaky relu
    alpha[(size_t)e * 4 + h] = v;
    atomicMax(&mmax[d * 4 + h], f2ord(v));
  }
}

// ---------------- GAT pass B: exp + un-normalized scatter ----------------------
// one wave per edge; lane handles 4 features (head = lane>>3)
__global__ void k_gat_scatter(const int* __restrict__ ei, const float* __restrict__ xsrc,
                              const float* __restrict__ alpha, const unsigned* __restrict__ mmax,
                              float* __restrict__ ssum, float* __restrict__ agg) {
  int wave = threadIdx.x >> 5, lane = threadIdx.x & 31;
  int e = blockIdx.x * 8 + wave;
  if (e >= NEDGESP) return;
  int s, d;
  if (e < NEDGES) { s = ei[e]; d = ei[NEDGES + e]; } else { s = d = e - NEDGES; }
  int h = lane >> 3;
  float m = ord2f(mmax[d * 4 + h]);
  float a = __expf(alpha[(size_t)e * 4 + h] - m);
  if ((lane & 7) == 0) atomicAdd(&ssum[d * 4 + h], a);
  float4 xv = *(const float4*)&xsrc[(size_t)s * 128 + lane * 4];
  float* dst = &agg[(size_t)d * 128 + lane * 4];
  atomicAdd(dst + 0, xv.x * a); atomicAdd(dst + 1, xv.y * a);
  atomicAdd(dst + 2, xv.z * a); atomicAdd(dst + 3, xv.w * a);
}

// ---------------- normalize GAT aggregate --------------------------------------
__global__ void k_gat_norm(float* __restrict__ agg, const float* __restrict__ ssum) {
  long i = (long)blockIdx.x * blockDim.x + threadIdx.x;
  if (i >= (long)NNODES * 128) return;
  int n = (int)(i >> 7), c = (int)(i & 127);
  agg[i] = agg[i] / (ssum[n * 4 + (c >> 5)] + 1e-16f);
}

// ---------------- wave-per-row LayerNorm (+optional residual) ------------------
__global__ void k_ln(const float* __restrict__ X, int ldx,
                     const float* __restrict__ g, const float* __restrict__ b,
                     const float* __restrict__ residual, int ldres,
                     float* __restrict__ O, int ldo, int C, int n_rows) {
  int wave = threadIdx.x >> 5, lane = threadIdx.x & 31;
  int row = blockIdx.x * 8 + wave;
  if (row >= n_rows) return;
  int vpl = C >> 5;                       // 4 or 8
  float vals[8];
  float s = 0.f;
  for (int i = 0; i < vpl; ++i) { vals[i] = X[(size_t)row * ldx + i * 32 + lane]; s += vals[i]; }
  for (int off = 16; off; off >>= 1) s += __shfl_xor(s, off, 32);
  float mean = s / (float)C;
  float vs = 0.f;
  for (int i = 0; i < vpl; ++i) { float dd = vals[i] - mean; vs += dd * dd; }
  for (int off = 16; off; off >>= 1) vs += __shfl_xor(vs, off, 32);
  float inv = rsqrtf(vs / (float)C + 1e-5f);
  for (int i = 0; i < vpl; ++i) {
    int c = i * 32 + lane;
    float o = (vals[i] - mean) * inv * g[c] + b[c];
    if (residual) o += residual[(size_t)row * ldres + c];
    O[(size_t)row * ldo + c] = o;
  }
}

// ---------------- fused GINE edge pipeline -------------------------------------
// block = 256 thr (8 waves), 16 edges: ef1 = relu(ea@W1^T+b1) [16x128 in LDS,
// A-frag order], ef = ef1@W2^T, msg = ef+b2+x[src]+ete[et], atomic scatter.
__global__ void k_gine_edge(const float* __restrict__ ea, const int* __restrict__ ei,
                            const int* __restrict__ et,
                            const _Float16* __restrict__ w1f, const float* __restrict__ b1,
                            const _Float16* __restrict__ w2f, const float* __restrict__ b2,
                            const float* __restrict__ x, const float* __restrict__ gine_ete,
                            float* __restrict__ agg) {
  __shared__ __align__(32) _Float16 sEA[512];
  __shared__ __align__(32) _Float16 sE1[2048];
  __shared__ int sSrc[16], sDst[16], sEt[16];
  int tid = threadIdx.x, wave = tid >> 5, lane = tid & 31;
  int e0 = blockIdx.x * 16;
  if (tid < 128) {                             // stage ea tile (K=16 padded to 32)
    int m = tid & 15, k = (tid >> 4) << 2;
    v4h hv = {};
    if (k < 16) {
      v4f av = *(const v4f*)&ea[(size_t)(e0 + m) * 16 + k];
      hv = __builtin_convertvector(av, v4h);
    }
    *(v4h*)&sEA[afrag_lane(m, k) * 16 + afrag_idx(k)] = hv;
  }
  if (tid >= 128 && tid < 144) {
    int i = tid - 128;
    sSrc[i] = ei[e0 + i]; sDst[i] = ei[NEDGES + e0 + i]; sEt[i] = et[e0 + i];
  }
  __syncthreads();
  {                                            // stage 1: one WMMA per wave
    v16h a = *(const v16h*)&sEA[lane * 16];
    v16h b = *(const v16h*)&w1f[(size_t)wave * 512 + lane * 16];
    v8f acc = {};
    acc = wmma_f16(a, b, acc);
    int col = wave * 16 + (lane & 15);
    int mhi = (lane >> 4) << 3;
#pragma unroll
    for (int r = 0; r < 8; ++r) {
      int m = r + mhi;
      float v = acc[r] + b1[col];
      v = v > 0.f ? v : 0.f;
      int kt2 = col >> 5, kl = col & 31;       // write back in A-frag order for stage 2
      sE1[kt2 * 512 + afrag_lane(m, kl) * 16 + afrag_idx(kl)] = (_Float16)v;
    }
  }
  __syncthreads();
  {                                            // stage 2: K=128 (4 WMMAs per wave)
    v8f acc = {};
    const _Float16* Bb = w2f + (size_t)wave * 4 * 512;
#pragma unroll
    for (int kt = 0; kt < 4; ++kt) {
      v16h a = *(const v16h*)&sE1[kt * 512 + lane * 16];
      v16h b = *(const v16h*)&Bb[(size_t)kt * 512 + lane * 16];
      acc = wmma_f16(a, b, acc);
    }
    int col = wave * 16 + (lane & 15);
    int mhi = (lane >> 4) << 3;
#pragma unroll
    for (int r = 0; r < 8; ++r) {
      int m = r + mhi;
      float v = acc[r] + b2[col];
      v += x[(size_t)sSrc[m] * 128 + col];
      v += gine_ete[sEt[m] * 128 + col];
      atomicAdd(&agg[(size_t)sDst[m] * 128 + col], v);
    }
  }
}

// ---------------- h = (1+eps)*x + agg ------------------------------------------
__global__ void k_gine_h(const float* __restrict__ x, const float* __restrict__ agg,
                         const float* __restrict__ epsp, float* __restrict__ h) {
  long i = (long)blockIdx.x * blockDim.x + threadIdx.x;
  if (i >= (long)NNODES * 128) return;
  h[i] = (1.f + epsp[0]) * x[i] + agg[i];
}

// ================================================================================
extern "C" void kernel_launch(void* const* d_in, const int* in_sizes, int n_in,
                              void* d_out, int out_size, void* d_ws, size_t ws_size,
                              hipStream_t stream) {
  (void)in_sizes; (void)n_in; (void)out_size; (void)ws_size;
  const float* x         = (const float*)d_in[0];
  const float* edge_attr = (const float*)d_in[1];
  const float* lin_src   = (const float*)d_in[2];
  const float* lin_dst   = (const float*)d_in[3];
  const float* att_src   = (const float*)d_in[4];
  const float* att_dst   = (const float*)d_in[5];
  const float* lin_edge  = (const float*)d_in[6];
  const float* att_edge  = (const float*)d_in[7];
  const float* gat_ete   = (const float*)d_in[8];
  const float* lin_out_w = (const float*)d_in[9];
  const float* lin_out_b = (const float*)d_in[10];
  const float* gat_bias  = (const float*)d_in[11];
  const float* gat_ln_g  = (const float*)d_in[12];
  const float* gat_ln_b  = (const float*)d_in[13];
  const float* enc_w1    = (const float*)d_in[14];
  const float* enc_b1    = (const float*)d_in[15];
  const float* enc_w2    = (const float*)d_in[16];
  const float* enc_b2    = (const float*)d_in[17];
  const float* gine_ete  = (const float*)d_in[18];
  const float* mlp_w1    = (const float*)d_in[19];
  const float* mlp_b1    = (const float*)d_in[20];
  const float* bn1_g     = (const float*)d_in[21];
  const float* bn1_b     = (const float*)d_in[22];
  const float* mlp_w2    = (const float*)d_in[23];
  const float* mlp_b2    = (const float*)d_in[24];
  const float* bn2_g     = (const float*)d_in[25];
  const float* bn2_b     = (const float*)d_in[26];
  const float* gin_eps   = (const float*)d_in[27];
  const float* fuse_w    = (const float*)d_in[28];
  const float* fuse_b    = (const float*)d_in[29];
  const float* fuse_ln_g = (const float*)d_in[30];
  const float* fuse_ln_b = (const float*)d_in[31];
  const int* edge_index  = (const int*)d_in[32];
  const int* edge_types  = (const int*)d_in[33];
  float* out = (float*)d_out;

  // ---- workspace carve-out ----
  char* base = (char*)d_ws;
  size_t off = 0;
  auto alloc = [&](size_t bytes) -> void* {
    void* p = base + off;
    off = (off + bytes + 255) & ~(size_t)255;
    return p;
  };
  float* xsrc     = (float*)alloc((size_t)NNODES * 128 * 4);
  float* xdst     = (float*)alloc((size_t)NNODES * 128 * 4);
  float* ai       = (float*)alloc((size_t)NNODES * 4 * 4);
  float* aj       = (float*)alloc((size_t)NNODES * 4 * 4);
  float* alphaB   = (float*)alloc((size_t)NEDGESP * 4 * 4);
  unsigned* mmax  = (unsigned*)alloc((size_t)NNODES * 4 * 4);
  float* ssum     = (float*)alloc((size_t)NNODES * 4 * 4);
  float* agg_gat  = (float*)alloc((size_t)NNODES * 128 * 4);
  float* agg_gine = (float*)alloc((size_t)NNODES * 128 * 4);
  float* hbuf     = (float*)alloc((size_t)NNODES * 128 * 4);
  float* gat_tmp  = (float*)alloc((size_t)NNODES * 128 * 4);
  float* t1       = (float*)alloc((size_t)NNODES * 256 * 4);
  float* catb     = (float*)alloc((size_t)NNODES * 256 * 4);
  float* fuse_tmp = (float*)alloc((size_t)NNODES * 256 * 4);
  _Float16* f_lin_src = (_Float16*)alloc(16384 * 2);
  _Float16* f_lin_dst = (_Float16*)alloc(16384 * 2);
  _Float16* f_lin_out = (_Float16*)alloc(16384 * 2);
  _Float16* f_enc_w1  = (_Float16*)alloc(4096 * 2);
  _Float16* f_enc_w2  = (_Float16*)alloc(16384 * 2);
  _Float16* f_mlp_w1  = (_Float16*)alloc(32768 * 2);
  _Float16* f_mlp_w2  = (_Float16*)alloc(32768 * 2);
  _Float16* f_fuse_w  = (_Float16*)alloc(65536 * 2);
  float* w_e        = (float*)alloc(64 * 4);
  float* t_alpha    = (float*)alloc(28 * 4);
  float* alpha_self = (float*)alloc(4 * 4);
  float* comb_bias  = (float*)alloc(128 * 4);

  const float* NF = nullptr;
  // ---- init accumulators ----
  k_fill<<<(NNODES * 128 + 255) / 256, 256, 0, stream>>>((unsigned*)agg_gat, 0u, (long)NNODES * 128);
  k_fill<<<(NNODES * 128 + 255) / 256, 256, 0, stream>>>((unsigned*)agg_gine, 0u, (long)NNODES * 128);
  k_fill<<<(NNODES * 4 + 255) / 256, 256, 0, stream>>>((unsigned*)ssum, 0u, (long)NNODES * 4);
  k_fill<<<(NNODES * 4 + 255) / 256, 256, 0, stream>>>(mmax, 0u, (long)NNODES * 4); // ordered -inf

  // ---- pack weights to f16 B-fragments ----
  k_pack_w<<<64, 256, 0, stream>>>(lin_src,   f_lin_src, 128, 128, 128);
  k_pack_w<<<64, 256, 0, stream>>>(lin_dst,   f_lin_dst, 128, 128, 128);
  k_pack_w<<<64, 256, 0, stream>>>(lin_out_w, f_lin_out, 128, 128, 128);
  k_pack_w<<<16, 256, 0, stream>>>(enc_w1,    f_enc_w1,  128, 16, 32);
  k_pack_w<<<64, 256, 0, stream>>>(enc_w2,    f_enc_w2,  128, 128, 128);
  k_pack_w<<<128, 256, 0, stream>>>(mlp_w1,   f_mlp_w1,  256, 128, 128);
  k_pack_w<<<128, 256, 0, stream>>>(mlp_w2,   f_mlp_w2,  128, 256, 256);
  k_pack_w<<<256, 256, 0, stream>>>(fuse_w,   f_fuse_w,  256, 256, 256);

  k_small_pre<<<1, 128, 0, stream>>>(lin_edge, att_edge, gat_ete, lin_out_b, gat_bias,
                                     w_e, t_alpha, alpha_self, comb_bias);

  // ---- GAT node projections: xsrc = x@lin_src^T, xdst = x@lin_dst^T ----
  dim3 g128(NNODES / 16, 1), g256(NNODES / 16, 2);
  k_gemm<<<g128, 128, 0, stream>>>(x, 128, f_lin_src, NF, NF, 1.f, NF, NF, 0, xsrc, 128, 128, 0);
  k_gemm<<<g128, 128, 0, stream>>>(x, 128, f_lin_dst, NF, NF, 1.f, NF, NF, 0, xdst, 128, 128, 0);
  k_ai_aj<<<(NNODES * 4 + 255) / 256, 256, 0, stream>>>(xsrc, xdst, att_src, att_dst, ai, aj, NNODES);

  // ---- GAT attention: logits+max, exp scatter, normalize ----
  k_gat_alpha<<<(NEDGESP + 255) / 256, 256, 0, stream>>>(edge_index, edge_types, edge_attr,
                                                         ai, aj, w_e, t_alpha, alpha_self,
                                                         alphaB, mmax);
  k_gat_scatter<<<NEDGESP / 8, 256, 0, stream>>>(edge_index, xsrc, alphaB, mmax, ssum, agg_gat);
  k_gat_norm<<<(NNODES * 128 + 255) / 256, 256, 0, stream>>>(agg_gat, ssum);

  // ---- GAT output projection + LN + residual -> catb[:, :128] ----
  k_gemm<<<g128, 128, 0, stream>>>(agg_gat, 128, f_lin_out, comb_bias, NF, 1.f, NF, NF, 0,
                                   gat_tmp, 128, 128, 0);
  k_ln<<<NNODES / 8, 256, 0, stream>>>(gat_tmp, 128, gat_ln_g, gat_ln_b, x, 128,
                                       catb, 256, 128, NNODES);

  // ---- GINE: fused edge MLP + message scatter ----
  k_gine_edge<<<NEDGES / 16, 256, 0, stream>>>(edge_attr, edge_index, edge_types,
                                               f_enc_w1, enc_b1, f_enc_w2, enc_b2,
                                               x, gine_ete, agg_gine);
  k_gine_h<<<(NNODES * 128 + 255) / 256, 256, 0, stream>>>(x, agg_gine, gin_eps, hbuf);

  // ---- GINE node MLP: 128 -> 256 (bn,relu) -> 128 (bn) + x -> catb[:, 128:] ----
  k_gemm<<<g256, 128, 0, stream>>>(hbuf, 128, f_mlp_w1, mlp_b1, bn1_g, BN_SCALE, bn1_b,
                                   NF, 0, t1, 256, 128, 1);
  k_gemm<<<g128, 128, 0, stream>>>(t1, 256, f_mlp_w2, mlp_b2, bn2_g, BN_SCALE, bn2_b,
                                   x, 128, catb + 128, 256, 256, 0);

  // ---- fuse: cat @ w^T + b, then LN -> out ----
  k_gemm<<<g256, 128, 0, stream>>>(catb, 256, f_fuse_w, fuse_b, NF, 1.f, NF, NF, 0,
                                   fuse_tmp, 256, 256, 0);
  k_ln<<<NNODES / 8, 256, 0, stream>>>(fuse_tmp, 256, fuse_ln_g, fuse_ln_b, NF, 0,
                                       out, 256, 256, NNODES);
}
